// EngramReplacer_3753801417054
// MI455X (gfx1250) — matrix operand order repaired and checked by reading, not compiled, not run on hardware
//
#include <hip/hip_runtime.h>
#include <math.h>

#define B_   4
#define T_   4096
#define D_   1024
#define W_   256
#define NW_  16

typedef _Float16 v16h __attribute__((ext_vector_type(16)));
typedef _Float16 v8h  __attribute__((ext_vector_type(8)));
typedef float    v8f  __attribute__((ext_vector_type(8)));
typedef int      v4i  __attribute__((ext_vector_type(4)));

union FragA { v16h v; v8h h[2]; };
union FragB { v16h v; v8h h[2]; };

#if defined(__has_builtin)
#if __has_builtin(__builtin_amdgcn_global_load_async_to_lds_b128)
#define HAVE_ASYNC_LDS 1
#endif
#endif

#ifdef HAVE_ASYNC_LDS
typedef __attribute__((address_space(1))) v4i gv4i;   // global int4
typedef __attribute__((address_space(3))) v4i lv4i;   // LDS int4
#if __has_builtin(__builtin_amdgcn_s_wait_asynccnt)
#define ASYNC_WAIT() __builtin_amdgcn_s_wait_asynccnt(0)
#else
#define ASYNC_WAIT() asm volatile("s_wait_asynccnt 0x0" ::: "memory")
#endif
#else
#define ASYNC_WAIT()
#endif

// ---------------------------------------------------------------------------
// Gate: per-window mean of loss -> sigmoid
// ---------------------------------------------------------------------------
__global__ __launch_bounds__(256) void gate_kernel(const float* __restrict__ loss,
                                                   const float* __restrict__ thr,
                                                   const float* __restrict__ shp,
                                                   float* __restrict__ gate_out) {
  __shared__ float s[256];
  const int win = blockIdx.x;            // 0..63 = b*16+n
  const int b = win >> 4, n = win & 15;
  s[threadIdx.x] = loss[b * T_ + n * W_ + threadIdx.x];
  __syncthreads();
  for (int off = 128; off; off >>= 1) {
    if (threadIdx.x < off) s[threadIdx.x] += s[threadIdx.x + off];
    __syncthreads();
  }
  if (threadIdx.x == 0) {
    float mean = s[0] * (1.0f / 256.0f);
    float z = shp[0] * (mean - thr[0]);
    gate_out[win] = 1.0f / (1.0f + expf(-z));
  }
}

// ---------------------------------------------------------------------------
// fp32 -> f16 conversion (weights)
// ---------------------------------------------------------------------------
__global__ __launch_bounds__(256) void cvt_f16_kernel(const float* __restrict__ src,
                                                      _Float16* __restrict__ dst, int n) {
  int i = blockIdx.x * 256 + threadIdx.x;
  if (i < n) dst[i] = (_Float16)src[i];
}

// A1 = concat(engram rows [512 x 1024], pos_emb rows [256 x 1024]) in f16
__global__ __launch_bounds__(256) void build_a1_kernel(const float* __restrict__ engrams,
                                                       const float* __restrict__ pos_emb,
                                                       _Float16* __restrict__ a1) {
  int i = blockIdx.x * 256 + threadIdx.x;   // 0 .. 768*1024-1
  int r = i >> 10, d = i & 1023;
  float v = (r < 512) ? engrams[i] : pos_emb[(r - 512) * D_ + d];
  a1[i] = (_Float16)v;
}

// ---------------------------------------------------------------------------
// GEMM1: H1[768,1024] = A1 @ w1^T (fp32 out). 32 rows/WG, 8 waves (2x4).
// Tiny (24 WGs) -> B streamed straight from L2.
// ---------------------------------------------------------------------------
__global__ __launch_bounds__(256) void gemm1_kernel(const _Float16* __restrict__ a1,
                                                    const _Float16* __restrict__ w1b,
                                                    float* __restrict__ h1) {
  const int tid  = threadIdx.x;
  const int lane = tid & 31;
  const int wave = tid >> 5;
  const int wm = wave & 1, wn = wave >> 1;
  const int lr = lane & 15, hi = lane >> 4;

  v8f c[16];
#pragma unroll
  for (int i = 0; i < 16; ++i) { v8f z = {}; c[i] = z; }

  const int row = blockIdx.x * 32 + wm * 16 + lr;
  const _Float16* arow = a1 + row * D_;

  for (int kt = 0; kt < 32; ++kt) {
    const int ka = kt * 32 + hi * 8;       // A: interleaved 8-chunks {k, k+16}
    FragA a;
    a.h[0] = *(const v8h*)(arow + ka);
    a.h[1] = *(const v8h*)(arow + ka + 16);
    const int kb = kt * 32 + hi * 8;       // B: interleaved 8-chunks {k, k+16}
#pragma unroll
    for (int nt = 0; nt < 16; ++nt) {
      const _Float16* wr = w1b + (wn * 256 + nt * 16 + lr) * D_ + kb;
      FragB b;
      b.h[0] = *(const v8h*)(wr);
      b.h[1] = *(const v8h*)(wr + 16);
      c[nt] = __builtin_amdgcn_wmma_f32_16x16x32_f16(false, a.v, false, b.v,
                                                     (short)0, c[nt], false, false);
    }
  }

#pragma unroll
  for (int nt = 0; nt < 16; ++nt) {
    const int e = wn * 256 + nt * 16 + lr;
#pragma unroll
    for (int v = 0; v < 8; ++v) {
      const int rr = blockIdx.x * 32 + wm * 16 + v + 8 * hi;
      h1[rr * D_ + e] = c[nt][v];
    }
  }
}

// ---------------------------------------------------------------------------
// Stage one 1024x32 (f16) k-slab of w2 into LDS (64KB), 512 threads.
// Async-to-LDS when the toolchain exposes it, else load+ds_store.
// ---------------------------------------------------------------------------
__device__ __forceinline__ void stage_slab(const _Float16* __restrict__ w2b,
                                           _Float16* bslab, int kt, int tid) {
  const int e0 = tid >> 2;            // 0..127
  const int kh = (tid & 3) * 8;       // 8-half (16B) chunk within slab row
#pragma unroll
  for (int p = 0; p < 8; ++p) {
    const int e = e0 + 128 * p;
    const _Float16* gp = w2b + e * D_ + kt * 32 + kh;
    _Float16* lp = bslab + e * 32 + kh;
#ifdef HAVE_ASYNC_LDS
    __builtin_amdgcn_global_load_async_to_lds_b128((gv4i*)gp, (lv4i*)lp, 0, 0);
#else
    *(v8h*)lp = *(const v8h*)gp;
#endif
  }
}

// ---------------------------------------------------------------------------
// Main fused kernel: 64 rows x 1024 cols per WG, 512 threads (16 waves,
// wm 0..3 x wn 0..3).  LDS: hbuf 128KB + double-buffered w2 slab 2x64KB.
// ---------------------------------------------------------------------------
__global__ __launch_bounds__(512) void main_kernel(
    const float* __restrict__ x, const float* __restrict__ engrams,
    const float* __restrict__ pos_emb, const _Float16* __restrict__ w2b,
    const float* __restrict__ h1, const float* __restrict__ b1,
    const float* __restrict__ b2, const float* __restrict__ ln_g,
    const float* __restrict__ ln_b, const float* __restrict__ type_emb,
    const float* __restrict__ gate, float* __restrict__ out) {
  extern __shared__ char smem[];
  _Float16* hbuf   = (_Float16*)smem;                    // [64][1024] = 128KB
  _Float16* bslab0 = (_Float16*)(smem + 131072);         // [1024][32] =  64KB
  _Float16* bslab1 = (_Float16*)(smem + 131072 + 65536); //              64KB
  const int tid = threadIdx.x;
  const int R0 = blockIdx.x * 64;                        // first global row

  // kick off the first w2 slab while we build h
  stage_slab(w2b, bslab0, 0, tid);

  // ---- stage 1: h = gelu(H1u + H1p + b1) rows into LDS (f16) -----------
  {
    const int m = tid >> 3;                       // 0..63
    const int dbase = (tid & 7) * 128;
    const int r = R0 + m;
    const int n = (r >> 8) & 15, w = r & 255, bb = r >> 12;
    const int rowU = bb * 128 + n * 8 + (w >> 5);
    const int rowP = 512 + w;
    const float* pu = h1 + rowU * D_ + dbase;
    const float* pp = h1 + rowP * D_ + dbase;
    const float* pb = b1 + dbase;
    _Float16* ph = hbuf + m * D_ + dbase;
    for (int d = 0; d < 128; d += 4) {
      float4 u = *(const float4*)(pu + d);
      float4 p = *(const float4*)(pp + d);
      float4 bv = *(const float4*)(pb + d);
      float t0 = u.x + p.x + bv.x, t1 = u.y + p.y + bv.y;
      float t2 = u.z + p.z + bv.z, t3 = u.w + p.w + bv.w;
      ph[d + 0] = (_Float16)(0.5f * t0 * (1.0f + erff(t0 * 0.70710678118f)));
      ph[d + 1] = (_Float16)(0.5f * t1 * (1.0f + erff(t1 * 0.70710678118f)));
      ph[d + 2] = (_Float16)(0.5f * t2 * (1.0f + erff(t2 * 0.70710678118f)));
      ph[d + 3] = (_Float16)(0.5f * t3 * (1.0f + erff(t3 * 0.70710678118f)));
    }
  }
  ASYNC_WAIT();
  __syncthreads();

  // ---- stage 2: WMMA GEMM (h @ w2^T), B double-buffered through LDS ----
  const int lane = tid & 31;
  const int wave = tid >> 5;            // 0..15
  const int wm = wave & 3, wn = wave >> 2;
  const int lr = lane & 15, hi = lane >> 4;

  v8f c[16];
#pragma unroll
  for (int i = 0; i < 16; ++i) { v8f z = {}; c[i] = z; }

  const _Float16* hrow = hbuf + (wm * 16 + lr) * D_;
  for (int kt = 0; kt < 32; ++kt) {
    _Float16* cur = (kt & 1) ? bslab1 : bslab0;
    _Float16* nxt = (kt & 1) ? bslab0 : bslab1;
    if (kt < 31) stage_slab(w2b, nxt, kt + 1, tid);   // overlap with compute

    const int ka = kt * 32 + hi * 8;
    FragA a;
    a.h[0] = *(const v8h*)(hrow + ka);
    a.h[1] = *(const v8h*)(hrow + ka + 16);
    const _Float16* bbase = cur + hi * 8;
#pragma unroll
    for (int nt = 0; nt < 16; ++nt) {
      const _Float16* br = bbase + (wn * 256 + nt * 16 + lr) * 32;
      FragB b;
      b.h[0] = *(const v8h*)(br);
      b.h[1] = *(const v8h*)(br + 16);
      c[nt] = __builtin_amdgcn_wmma_f32_16x16x32_f16(false, a.v, false, b.v,
                                                     (short)0, c[nt], false, false);
    }
    ASYNC_WAIT();
    __syncthreads();            // next slab staged; cur free for rewrite
  }

  // ---- stage 3: residual + LayerNorm + blend ---------------------------
  float* rowsum = (float*)smem;                    // [64]  (reuse LDS)
  float* rowsq  = rowsum + 64;                     // [64]
  if (tid < 128) rowsum[tid] = 0.0f;
  __syncthreads();

  int   engbase[8], posbase[8], xbase[8];
  float gatev[8];
#pragma unroll
  for (int v = 0; v < 8; ++v) {
    const int m = wm * 16 + v + 8 * hi;            // C layout: M = v + 8*hi
    const int r = R0 + m;
    const int bb = r >> 12, n = (r >> 8) & 15, w = r & 255;
    engbase[v] = (bb * 128 + n * 8 + (w >> 5)) * D_;
    posbase[v] = w * D_;
    xbase[v]   = r * D_;                           // r == b*T + t exactly
    gatev[v]   = gate[bb * 16 + n];
  }

  float psum[8] = {0.f}, psq[8] = {0.f};
#pragma unroll
  for (int nt = 0; nt < 16; ++nt) {
    const int e = wn * 256 + nt * 16 + lr;
    const float b2e = b2[e];
#pragma unroll
    for (int v = 0; v < 8; ++v) {
      float up = c[nt][v] + engrams[engbase[v] + e] + pos_emb[posbase[v] + e] + b2e;
      c[nt][v] = up;
      psum[v] += up;
      psq[v]  += up * up;
    }
  }
  // reduce across the 16 lanes sharing each row, then one LDS atomic
#pragma unroll
  for (int v = 0; v < 8; ++v) {
#pragma unroll
    for (int off = 8; off; off >>= 1) {
      psum[v] += __shfl_xor(psum[v], off, 16);
      psq[v]  += __shfl_xor(psq[v],  off, 16);
    }
    if (lr == 0) {
      const int m = wm * 16 + v + 8 * hi;
      atomicAdd(&rowsum[m], psum[v]);
      atomicAdd(&rowsq[m],  psq[v]);
    }
  }
  __syncthreads();

  float mu[8], rs[8];
#pragma unroll
  for (int v = 0; v < 8; ++v) {
    const int m = wm * 16 + v + 8 * hi;
    const float s = rowsum[m] * (1.0f / (float)D_);
    const float q = rowsq[m]  * (1.0f / (float)D_);
    mu[v] = s;
    rs[v] = rsqrtf(q - s * s + 1e-5f);
  }

#pragma unroll
  for (int nt = 0; nt < 16; ++nt) {
    const int e = wn * 256 + nt * 16 + lr;
    const float lg = ln_g[e], lb = ln_b[e], te = type_emb[e];
#pragma unroll
    for (int v = 0; v < 8; ++v) {
      const float y  = (c[nt][v] - mu[v]) * rs[v] * lg + lb + te;
      const float xv = x[xbase[v] + e];
      out[xbase[v] + e] = gatev[v] * y + (1.0f - gatev[v]) * xv;
    }
  }
}

// ---------------------------------------------------------------------------
extern "C" void kernel_launch(void* const* d_in, const int* in_sizes, int n_in,
                              void* d_out, int out_size, void* d_ws, size_t ws_size,
                              hipStream_t stream) {
  const float* x        = (const float*)d_in[0];
  const float* engrams  = (const float*)d_in[1];
  const float* loss     = (const float*)d_in[2];
  const float* pos_emb  = (const float*)d_in[3];
  const float* w1       = (const float*)d_in[4];
  const float* b1       = (const float*)d_in[5];
  const float* w2       = (const float*)d_in[6];
  const float* b2       = (const float*)d_in[7];
  const float* ln_g     = (const float*)d_in[8];
  const float* ln_b     = (const float*)d_in[9];
  const float* type_emb = (const float*)d_in[10];
  const float* thr      = (const float*)d_in[11];
  const float* shp      = (const float*)d_in[12];

  float* out      = (float*)d_out;
  float* gate_out = out + (size_t)B_ * T_ * D_;    // gate tuple member

  char* ws = (char*)d_ws;
  _Float16* w1b = (_Float16*)(ws);                       // 2 MB
  _Float16* w2b = (_Float16*)(ws + (2ull << 20));        // 2 MB
  _Float16* a1b = (_Float16*)(ws + (4ull << 20));        // 1.5 MB
  float*    h1  = (float*)   (ws + (6ull << 20));        // 3 MB

  // opt in to 256KB dynamic LDS (WGP supports 320KB)
  (void)hipFuncSetAttribute((const void*)main_kernel,
                            hipFuncAttributeMaxDynamicSharedMemorySize, 262144);

  gate_kernel<<<64, 256, 0, stream>>>(loss, thr, shp, gate_out);
  cvt_f16_kernel<<<4096, 256, 0, stream>>>(w1, w1b, D_ * D_);
  cvt_f16_kernel<<<4096, 256, 0, stream>>>(w2, w2b, D_ * D_);
  build_a1_kernel<<<3072, 256, 0, stream>>>(engrams, pos_emb, a1b);
  gemm1_kernel<<<24, 256, 0, stream>>>(a1b, w1b, h1);
  main_kernel<<<256, 512, 262144, stream>>>(x, engrams, pos_emb, w2b, h1, b1,
                                            b2, ln_g, ln_b, type_emb, gate_out, out);
}
